// VibyBlock_81432579932349
// MI455X (gfx1250) — matrix operand order repaired and checked by reading, not compiled, not run on hardware
//
#include <hip/hip_runtime.h>
#include <stdint.h>
#include <stddef.h>

// ---- problem dims ----
#define BB   2
#define SSQ  2048
#define DDm  1024
#define HHn  16
#define KVH  4
#define HDD  64
#define IIm  3584
#define EEx  4
#define WINW 512

typedef __bf16 bf16_t;
typedef __attribute__((ext_vector_type(16))) __bf16 v16bf;
typedef __attribute__((ext_vector_type(8)))  __bf16 v8bf;
typedef __attribute__((ext_vector_type(8)))  float  v8f;
typedef __attribute__((ext_vector_type(4)))  unsigned int u32x4;
typedef __attribute__((ext_vector_type(8)))  int i32x8;
typedef __attribute__((ext_vector_type(4)))  int i32x4;

__device__ __forceinline__ float  bf2f(bf16_t x){ return (float)x; }
__device__ __forceinline__ bf16_t f2bf(float x){ return (bf16_t)x; }

enum { EPI_F32=0, EPI_BF16=1, EPI_GCLIP=2, EPI_UCLIP=3, EPI_ADDF32=4, EPI_SACC=5 };

// B-tile staging geometry: 32 K-rows x 64 cols bf16, padded row stride 144B
// (TDM pad: 4 DWORDs after every 32 DWORDs -> avoids 64-bank conflicts on
//  the per-lane row reads, and keeps rows 16B aligned for ds_load_b128).
#define BROW_BYTES 144
#define BSTG (32 * BROW_BYTES)   // 4608 bytes per buffer

#if defined(__AMDGCN__) && __has_builtin(__builtin_amdgcn_tensor_load_to_lds)
#define USE_TDM 1
#endif

#ifdef USE_TDM
// Tensor Data Mover: async DMA of a 2D tile (32 rows x 64 bf16, row stride
// n_stride elements) from global into LDS at byte offset lds_off.
// D# layout per CDNA5 ISA (groups 0 and 1; 2D tensor).
__device__ __forceinline__ void tdm_stage_b(uint32_t lds_off, const bf16_t* g,
                                            uint32_t n_stride)
{
    uint64_t ga = (uint64_t)(uintptr_t)g;
    u32x4 g0;
    g0.x = 1u;                                                // count=1 (valid), user mode
    g0.y = lds_off;                                           // lds_addr[63:32]
    g0.z = (uint32_t)ga;                                      // global_addr[95:64]
    g0.w = (uint32_t)((ga >> 32) & 0x01ffffffu) | (2u << 30); // addr[120:96] | type=2
    i32x8 g1;
    g1[0] = (int)((1u << 16)      // data_size = 2 bytes
                | (1u << 20)      // pad_enable
                | (4u << 22)      // pad_interval: code 4 = 32 DWORDs (128B row)
                | (3u << 25));    // pad_amount:   code 3 = 4 DWORDs (16B)
    g1[1] = (int)(64u << 16);     // tensor_dim0 = 64 (lo16 at bits 63:48)
    g1[2] = (int)(32u << 16);     // tensor_dim0 hi = 0 | tensor_dim1 = 32
    g1[3] = (int)(64u << 16);     // tensor_dim1 hi = 0 | tile_dim0 = 64
    g1[4] = (int)32u;             // tile_dim1 = 32, tile_dim2 = 0
    g1[5] = (int)n_stride;        // tensor_dim0_stride lo32 (elements)
    g1[6] = 0;                    // stride hi16 | tensor_dim1_stride lo16
    g1[7] = 0;
    i32x4 z4 = {0, 0, 0, 0};
#if __clang_major__ >= 23
    i32x8 z8 = {0, 0, 0, 0, 0, 0, 0, 0};
    __builtin_amdgcn_tensor_load_to_lds(g0, g1, z4, z4, z8, 0);
#else
    __builtin_amdgcn_tensor_load_to_lds(g0, g1, z4, z4, 0);
#endif
}
#else
// Fallback: cooperative staging with the same padded layout.
__device__ __forceinline__ void coop_stage_b(char* dst, const bf16_t* g, int N)
{
    for (int idx = threadIdx.x; idx < 512; idx += 256) {
        int r = idx >> 4;          // 0..31
        int c = (idx & 15) << 2;   // 0..60 step 4
        *(uint64_t*)(dst + r * BROW_BYTES + c * 2) =
            *(const uint64_t*)(g + (size_t)r * N + c);
    }
}
#endif

// ============================================================
// Weight convert + transpose: Bt[k][n] = bf16(W[n][k]),  W is (N,K) row-major
// ============================================================
__global__ void k_conv_t(const float* __restrict__ W, bf16_t* __restrict__ Bt,
                         int N, int K)
{
    size_t i = (size_t)blockIdx.x * 256 + threadIdx.x;
    size_t total = (size_t)N * K;
    if (i < total) {
        int n = (int)(i % N);
        int k = (int)(i / N);
        Bt[i] = f2bf(W[(size_t)n * K + k]);
    }
}

// ============================================================
// RMSNorm over rows of length D, fp32 in -> bf16 out. 1 wave / row.
// ============================================================
__global__ void k_rmsnorm(const float* __restrict__ X, const float* __restrict__ W,
                          bf16_t* __restrict__ Out, int D)
{
    int row  = blockIdx.x;
    int lane = threadIdx.x;
    const float* x = X + (size_t)row * D;
    float ss = 0.f;
    for (int i = lane; i < D; i += 32) { float v = x[i]; ss += v * v; }
    #pragma unroll
    for (int m = 16; m > 0; m >>= 1) ss += __shfl_xor(ss, m, 32);
    float r = rsqrtf(ss / (float)D + 1e-5f);
    bf16_t* o = Out + (size_t)row * D;
    for (int i = lane; i < D; i += 32) o[i] = f2bf(x[i] * r * W[i]);
}

// ============================================================
// WMMA GEMM: C(MxN) = A(MxK,bf16) * Bt(KxN,bf16), f32 accumulate.
// Block tile 128x64, 8 waves x (16x64 strip). The 32x64 B tile for each
// k-step is staged into LDS once per block (TDM async DMA, double buffered,
// completion via TENSORcnt) and shared by all 8 waves via ds loads.
// A-frag: lane = (M row | k-half) per ISA 16-bit A table.
// B-frag: lane = N column, elements = K rows, per ISA B table.
// ============================================================
__global__ __launch_bounds__(256)
void k_gemm(const bf16_t* __restrict__ A, const bf16_t* __restrict__ Bt,
            void* __restrict__ Out, const float* __restrict__ Res,
            const float* __restrict__ Scale, int scale_stride,
            int M, int N, int K, int epi)
{
    __shared__ char smemB[2 * BSTG];

    int wave = threadIdx.x >> 5;
    int lane = threadIdx.x & 31;
    int lh   = lane >> 4;      // 0/1 : k-half for A, M+8 for C
    int ll   = lane & 15;
    int row0 = blockIdx.y * 128 + wave * 16;
    int col0 = blockIdx.x * 64;

    v8f c0 = {0,0,0,0,0,0,0,0}, c1 = c0, c2 = c0, c3 = c0;

    const bf16_t* arow = A + (size_t)(row0 + ll) * K + 8 * lh;
    const int niter = K >> 5;

#ifdef USE_TDM
    uint32_t ldsbase = (uint32_t)(uintptr_t)(void*)smemB;
    if (threadIdx.x < 32)
        tdm_stage_b(ldsbase, Bt + col0, (uint32_t)N);
#else
    coop_stage_b(smemB, Bt + col0, N);
#endif

    for (int i = 0; i < niter; ++i) {
        const int kk = i << 5;
        __syncthreads();   // all waves done reading the buffer we stage next
#ifdef USE_TDM
        if (threadIdx.x < 32) {
            if (i + 1 < niter)
                tdm_stage_b(ldsbase + (uint32_t)(((i + 1) & 1) * BSTG),
                            Bt + (size_t)(kk + 32) * N + col0, (uint32_t)N);
            if (i + 1 < niter) __builtin_amdgcn_s_wait_tensorcnt((short)1);
            else               __builtin_amdgcn_s_wait_tensorcnt((short)0);
        }
#else
        if (i + 1 < niter)
            coop_stage_b(smemB + ((i + 1) & 1) * BSTG,
                         Bt + (size_t)(kk + 32) * N + col0, N);
#endif
        __syncthreads();   // current buffer ready for every wave

        union { v16bf v; v8bf h2[2]; } af;
        af.h2[0] = *(const v8bf*)(arow + kk);        // K = kk+8lh .. +7
        af.h2[1] = *(const v8bf*)(arow + kk + 16);   // K = kk+16+8lh .. +7

        const char* bcol = smemB + (i & 1) * BSTG + ll * 2;  // lane = N column, elements = K rows
        union { v16bf v; v8bf h2[2]; } b0, b1, b2, b3;
        #pragma unroll
        for (int s = 0; s < 16; ++s) {
            int kr = (s < 8) ? (8 * lh + s) : (16 + 8 * lh + (s - 8));
            const char* bp = bcol + kr * BROW_BYTES;
            b0.v[s] = *(const bf16_t*)(bp);
            b1.v[s] = *(const bf16_t*)(bp + 32);
            b2.v[s] = *(const bf16_t*)(bp + 64);
            b3.v[s] = *(const bf16_t*)(bp + 96);
        }

        c0 = __builtin_amdgcn_wmma_f32_16x16x32_bf16(false, af.v, false, b0.v, (short)0, c0, false, false);
        c1 = __builtin_amdgcn_wmma_f32_16x16x32_bf16(false, af.v, false, b1.v, (short)0, c1, false, false);
        c2 = __builtin_amdgcn_wmma_f32_16x16x32_bf16(false, af.v, false, b2.v, (short)0, c2, false, false);
        c3 = __builtin_amdgcn_wmma_f32_16x16x32_bf16(false, af.v, false, b3.v, (short)0, c3, false, false);
    }

    #pragma unroll
    for (int v = 0; v < 8; ++v) {
        int row = row0 + v + 8 * lh;
        float vals[4] = { c0[v], c1[v], c2[v], c3[v] };
        #pragma unroll
        for (int t = 0; t < 4; ++t) {
            int col = col0 + 16 * t + ll;
            size_t idx = (size_t)row * N + col;
            float x = vals[t];
            switch (epi) {
                case EPI_F32:   ((float*) Out)[idx] = x; break;
                case EPI_BF16:  ((bf16_t*)Out)[idx] = f2bf(x); break;
                case EPI_GCLIP: ((bf16_t*)Out)[idx] = f2bf(fminf(x, 7.0f)); break;
                case EPI_UCLIP: ((bf16_t*)Out)[idx] = f2bf(fminf(fmaxf(x, -7.0f), 7.0f)); break;
                case EPI_ADDF32:((float*) Out)[idx] = x + Res[idx]; break;
                case EPI_SACC: {
                    float s = Scale ? Scale[(size_t)row * scale_stride] : 1.0f;
                    ((float*)Out)[idx] += s * x;
                } break;
            }
        }
    }
}

// ============================================================
// Q path: head RMSNorm + RoPE + 1/sqrt(HD) scale -> Qb[b][h][s][hd] bf16
// ============================================================
__global__ void k_rope_q(const float* __restrict__ Qraw, const float* __restrict__ Wn,
                         const float* __restrict__ Cos, const float* __restrict__ Sin,
                         bf16_t* __restrict__ Qb, float qscale)
{
    int idx  = blockIdx.x;                 // (b*SSQ + s)*HHn + h
    int h    = idx % HHn;
    int s    = (idx / HHn) % SSQ;
    int b    =  idx / (HHn * SSQ);
    int lane = threadIdx.x;
    const float* q = Qraw + (size_t)idx * HDD;
    float e0 = q[lane], e1 = q[lane + 32];
    float ss = e0 * e0 + e1 * e1;
    #pragma unroll
    for (int m = 16; m > 0; m >>= 1) ss += __shfl_xor(ss, m, 32);
    float r = rsqrtf(ss / (float)HDD + 1e-5f);
    e0 = e0 * r * Wn[lane];
    e1 = e1 * r * Wn[lane + 32];
    float c0 = Cos[s * HDD + lane], c1 = Cos[s * HDD + lane + 32];
    float s0 = Sin[s * HDD + lane], s1 = Sin[s * HDD + lane + 32];
    float o0 = e0 * c0 - e1 * s0;          // rot_half: first half gets -x[hi]
    float o1 = e1 * c1 + e0 * s1;
    bf16_t* o = Qb + (((size_t)(b * HHn + h)) * SSQ + s) * HDD;
    o[lane]      = f2bf(o0 * qscale);
    o[lane + 32] = f2bf(o1 * qscale);
}

// ============================================================
// K path: head RMSNorm + RoPE -> KT[b][kv][hd][s] bf16 (transposed for B-frags)
// ============================================================
__global__ void k_rope_k(const float* __restrict__ Kraw, const float* __restrict__ Wn,
                         const float* __restrict__ Cos, const float* __restrict__ Sin,
                         bf16_t* __restrict__ KT)
{
    int idx  = blockIdx.x;                 // (b*SSQ + s)*KVH + kv
    int kv   = idx % KVH;
    int s    = (idx / KVH) % SSQ;
    int b    =  idx / (KVH * SSQ);
    int lane = threadIdx.x;
    const float* k = Kraw + (size_t)idx * HDD;
    float e0 = k[lane], e1 = k[lane + 32];
    float ss = e0 * e0 + e1 * e1;
    #pragma unroll
    for (int m = 16; m > 0; m >>= 1) ss += __shfl_xor(ss, m, 32);
    float r = rsqrtf(ss / (float)HDD + 1e-5f);
    e0 = e0 * r * Wn[lane];
    e1 = e1 * r * Wn[lane + 32];
    float c0 = Cos[s * HDD + lane], c1 = Cos[s * HDD + lane + 32];
    float s0 = Sin[s * HDD + lane], s1 = Sin[s * HDD + lane + 32];
    float o0 = e0 * c0 - e1 * s0;
    float o1 = e1 * c1 + e0 * s1;
    bf16_t* base = KT + ((size_t)(b * KVH + kv) * HDD) * SSQ;
    base[(size_t)lane        * SSQ + s] = f2bf(o0);
    base[(size_t)(lane + 32) * SSQ + s] = f2bf(o1);
}

// V: (B,S,KV,HD) f32 -> Vb[b][kv][s][hd] bf16
__global__ void k_vconv(const float* __restrict__ Vraw, bf16_t* __restrict__ Vb)
{
    size_t i = (size_t)blockIdx.x * 256 + threadIdx.x;
    size_t total = (size_t)BB * SSQ * KVH * HDD;
    if (i >= total) return;
    int hd = (int)(i % HDD);
    int kv = (int)((i / HDD) % KVH);
    int s  = (int)((i / ((size_t)HDD * KVH)) % SSQ);
    int b  = (int)( i / ((size_t)HDD * KVH * SSQ));
    Vb[((size_t)(b * KVH + kv) * SSQ + s) * HDD + hd] = f2bf(Vraw[i]);
}

// ============================================================
// Flash attention with WMMA. 1 wave per (b, h, 16-query tile).
// Scores via Q(A) x KT(B), P staged through LDS to A-layout, P x V via WMMA.
// Sink logit folded into the final softmax rescale.
// ============================================================
__global__ __launch_bounds__(32)
void k_attn(const bf16_t* __restrict__ Qb, const bf16_t* __restrict__ KT,
            const bf16_t* __restrict__ Vb, const float* __restrict__ Sinks,
            bf16_t* __restrict__ Ob)
{
    __shared__ bf16_t P[16 * 32];

    int blk  = blockIdx.x;
    int qt   = blk & (SSQ / 16 - 1);       // 128 tiles
    int h    = (blk >> 7) & (HHn - 1);
    int b    =  blk >> 11;
    int kv   = h >> 2;                     // NREP = 4
    int lane = threadIdx.x;
    int lh   = lane >> 4, ll = lane & 15;
    int q0   = qt * 16;

    // Q A-fragments for hd chunks [0,32) and [32,64)
    const bf16_t* qrow = Qb + (((size_t)(b * HHn + h)) * SSQ + q0 + ll) * HDD + 8 * lh;
    union { v16bf v; v8bf h2[2]; } a0, a1;
    a0.h2[0] = *(const v8bf*)(qrow);       a0.h2[1] = *(const v8bf*)(qrow + 16);
    a1.h2[0] = *(const v8bf*)(qrow + 32);  a1.h2[1] = *(const v8bf*)(qrow + 48);

    const bf16_t* ktbase = KT + ((size_t)(b * KVH + kv) * HDD) * SSQ;   // [hd][s]
    const bf16_t* vbase  = Vb + ((size_t)(b * KVH + kv)) * SSQ * HDD;   // [s][hd]

    v8f o0 = {0,0,0,0,0,0,0,0}, o1 = o0, o2 = o0, o3 = o0;
    float mrow[8], lrow[8];
    #pragma unroll
    for (int v = 0; v < 8; ++v) { mrow[v] = -1e30f; lrow[v] = 0.f; }

    int lo = q0 - (WINW - 1); if (lo < 0) lo = 0; lo &= ~31;
    int hi = q0 + 15;

    for (int kc = lo; kc <= hi; kc += 32) {
        // ---- scores: two 16-key tiles, K-dim = hd (2 chunks of 32) ----
        v8f s0 = {0,0,0,0,0,0,0,0}, s1 = s0;
        {
            v16bf b00, b01, b10, b11;
            #pragma unroll
            for (int s = 0; s < 16; ++s) {
                int hd = (s < 8) ? (8 * lh + s) : (16 + 8 * lh + (s - 8));
                const bf16_t* kp0 = ktbase + (size_t)hd * SSQ + kc + ll;
                const bf16_t* kp1 = kp0 + (size_t)32 * SSQ;
                b00[s] = kp0[0]; b01[s] = kp0[16];
                b10[s] = kp1[0]; b11[s] = kp1[16];
            }
            s0 = __builtin_amdgcn_wmma_f32_16x16x32_bf16(false, a0.v, false, b00, (short)0, s0, false, false);
            s0 = __builtin_amdgcn_wmma_f32_16x16x32_bf16(false, a1.v, false, b10, (short)0, s0, false, false);
            s1 = __builtin_amdgcn_wmma_f32_16x16x32_bf16(false, a0.v, false, b01, (short)0, s1, false, false);
            s1 = __builtin_amdgcn_wmma_f32_16x16x32_bf16(false, a1.v, false, b11, (short)0, s1, false, false);
        }

        // ---- mask + online softmax (rows live in 16-lane halves) ----
        float p0[8], p1[8];
        #pragma unroll
        for (int v = 0; v < 8; ++v) {
            int qr  = q0 + v + 8 * lh;
            int k0i = kc + ll;
            int k1i = kc + 16 + ll;
            float x0 = ((k0i <= qr) && (k0i > qr - WINW)) ? s0[v] : -1e30f;
            float x1 = ((k1i <= qr) && (k1i > qr - WINW)) ? s1[v] : -1e30f;
            float cm = fmaxf(x0, x1);
            #pragma unroll
            for (int mm = 8; mm > 0; mm >>= 1) cm = fmaxf(cm, __shfl_xor(cm, mm, 32));
            float nm    = fmaxf(mrow[v], cm);
            float alpha = __expf(mrow[v] - nm);
            float e0 = __expf(x0 - nm);
            float e1 = __expf(x1 - nm);
            float rs = e0 + e1;
            #pragma unroll
            for (int mm = 8; mm > 0; mm >>= 1) rs += __shfl_xor(rs, mm, 32);
            lrow[v] = lrow[v] * alpha + rs;
            mrow[v] = nm;
            p0[v] = e0; p1[v] = e1;
            o0[v] *= alpha; o1[v] *= alpha; o2[v] *= alpha; o3[v] *= alpha;
        }

        // ---- stage P in LDS (C-layout -> [q][key] matrix) ----
        __syncthreads();
        #pragma unroll
        for (int v = 0; v < 8; ++v) {
            int r = v + 8 * lh;
            P[r * 32 + ll]      = f2bf(p0[v]);
            P[r * 32 + 16 + ll] = f2bf(p1[v]);
        }
        __syncthreads();

        // ---- reload P as A-fragment (16x32 over keys) ----
        union { v16bf v; bf16_t e[16]; } pA;
        #pragma unroll
        for (int v = 0; v < 8; ++v) {
            int k = 2 * (v & 3) + 16 * (v >> 2) + 8 * lh;
            pA.e[2 * v]     = P[ll * 32 + k];
            pA.e[2 * v + 1] = P[ll * 32 + k + 1];
        }

        // ---- V B-fragments (lane = hd column, elements = keys) ----
        v16bf vb0, vb1, vb2, vb3;
        #pragma unroll
        for (int s = 0; s < 16; ++s) {
            int kr = (s < 8) ? (8 * lh + s) : (16 + 8 * lh + (s - 8));
            const bf16_t* vp = vbase + (size_t)(kc + kr) * HDD + ll;
            vb0[s] = vp[0]; vb1[s] = vp[16]; vb2[s] = vp[32]; vb3[s] = vp[48];
        }
        o0 = __builtin_amdgcn_wmma_f32_16x16x32_bf16(false, pA.v, false, vb0, (short)0, o0, false, false);
        o1 = __builtin_amdgcn_wmma_f32_16x16x32_bf16(false, pA.v, false, vb1, (short)0, o1, false, false);
        o2 = __builtin_amdgcn_wmma_f32_16x16x32_bf16(false, pA.v, false, vb2, (short)0, o2, false, false);
        o3 = __builtin_amdgcn_wmma_f32_16x16x32_bf16(false, pA.v, false, vb3, (short)0, o3, false, false);
    }

    // ---- sink logit + final normalize ----
    float snk = Sinks[h];
    #pragma unroll
    for (int v = 0; v < 8; ++v) {
        float nm    = fmaxf(mrow[v], snk);
        float alpha = __expf(mrow[v] - nm);
        float l     = lrow[v] * alpha + __expf(snk - nm);
        float sc    = alpha / l;
        o0[v] *= sc; o1[v] *= sc; o2[v] *= sc; o3[v] *= sc;
    }

    // ---- store to Ob[b][s][h*64 + hd] (bf16, feeds wo GEMM) ----
    #pragma unroll
    for (int v = 0; v < 8; ++v) {
        int row = q0 + v + 8 * lh;
        bf16_t* dst = Ob + ((size_t)(b * SSQ) + row) * (HHn * HDD) + h * HDD;
        dst[ll]      = f2bf(o0[v]);
        dst[16 + ll] = f2bf(o1[v]);
        dst[32 + ll] = f2bf(o2[v]);
        dst[48 + ll] = f2bf(o3[v]);
    }
}

// ============================================================
// Router: logits = m . gate_w^T (E=4), softmax, top-2, renormalize -> Wd[t][e]
// ============================================================
__global__ void k_gate(const bf16_t* __restrict__ Mb, const float* __restrict__ Gw,
                       float* __restrict__ Wd)
{
    int t    = blockIdx.x;
    int lane = threadIdx.x;
    const bf16_t* m = Mb + (size_t)t * DDm;
    float acc[EEx] = {0.f, 0.f, 0.f, 0.f};
    for (int i = lane; i < DDm; i += 32) {
        float mv = bf2f(m[i]);
        acc[0] += mv * Gw[i];
        acc[1] += mv * Gw[DDm + i];
        acc[2] += mv * Gw[2 * DDm + i];
        acc[3] += mv * Gw[3 * DDm + i];
    }
    #pragma unroll
    for (int e = 0; e < EEx; ++e)
        #pragma unroll
        for (int mm = 16; mm > 0; mm >>= 1) acc[e] += __shfl_xor(acc[e], mm, 32);

    if (lane == 0) {
        float mx = fmaxf(fmaxf(acc[0], acc[1]), fmaxf(acc[2], acc[3]));
        float ex[EEx], s = 0.f;
        for (int e = 0; e < EEx; ++e) { ex[e] = __expf(acc[e] - mx); s += ex[e]; }
        for (int e = 0; e < EEx; ++e) ex[e] /= s;
        int i0 = 0;
        for (int e = 1; e < EEx; ++e) if (ex[e] > ex[i0]) i0 = e;
        int i1 = -1;
        for (int e = 0; e < EEx; ++e) { if (e == i0) continue; if (i1 < 0 || ex[e] > ex[i1]) i1 = e; }
        float tws = ex[i0] + ex[i1] + 1e-20f;
        float w[EEx] = {0.f, 0.f, 0.f, 0.f};
        w[i0] = ex[i0] / tws;
        w[i1] = ex[i1] / tws;
        for (int e = 0; e < EEx; ++e) Wd[(size_t)t * EEx + e] = w[e];
    }
}

// h = silu(g) * u  (elementwise, bf16). Out may alias G.
__global__ void k_silu_mul(const bf16_t* G, const bf16_t* U, bf16_t* Out, size_t n)
{
    size_t i = (size_t)blockIdx.x * 256 + threadIdx.x;
    if (i < n) {
        float g = bf2f(G[i]);
        float u = bf2f(U[i]);
        float s = g / (1.f + __expf(-g));
        Out[i] = f2bf(s * u);
    }
}

__global__ void k_zero(float* __restrict__ Y, size_t n)
{
    size_t i = (size_t)blockIdx.x * 256 + threadIdx.x;
    if (i < n) Y[i] = 0.f;
}

__global__ void k_addout(const float* __restrict__ H1, const float* __restrict__ Y,
                         float* __restrict__ Out, size_t n)
{
    size_t i = (size_t)blockIdx.x * 256 + threadIdx.x;
    if (i < n) Out[i] = H1[i] + Y[i];
}

// ============================================================
// Host orchestration
// ============================================================
extern "C" void kernel_launch(void* const* d_in, const int* in_sizes, int n_in,
                              void* d_out, int out_size, void* d_ws, size_t ws_size,
                              hipStream_t stream)
{
    (void)in_sizes; (void)n_in; (void)out_size; (void)ws_size;

    const float* x      = (const float*)d_in[0];
    const float* cosT   = (const float*)d_in[1];
    const float* sinT   = (const float*)d_in[2];
    const float* anw    = (const float*)d_in[3];
    const float* pnw    = (const float*)d_in[4];
    const float* qnw    = (const float*)d_in[5];
    const float* knw    = (const float*)d_in[6];
    const float* wq     = (const float*)d_in[7];
    const float* wk     = (const float*)d_in[8];
    const float* wv     = (const float*)d_in[9];
    const float* wo     = (const float*)d_in[10];
    const float* sinks  = (const float*)d_in[11];
    const float* gatew  = (const float*)d_in[12];
    const float* egw    = (const float*)d_in[13];
    const float* euw    = (const float*)d_in[14];
    const float* edw    = (const float*)d_in[15];
    const float* sgw    = (const float*)d_in[16];
    const float* suw    = (const float*)d_in[17];
    const float* sdw    = (const float*)d_in[18];
    float* out = (float*)d_out;

    const size_t T = (size_t)BB * SSQ;            // 4096 tokens
    const int NQ = HHn * HDD;                     // 1024
    const int NK = KVH * HDD;                     // 256

    // ---- workspace carve-up ----
    char* p = (char*)d_ws;
    auto take = [&](size_t bytes) -> void* {
        void* r = (void*)p;
        p += (bytes + 255) & ~(size_t)255;
        return r;
    };
    bf16_t* wqt  = (bf16_t*)take((size_t)DDm * NQ  * 2);
    bf16_t* wkt  = (bf16_t*)take((size_t)DDm * NK  * 2);
    bf16_t* wvt  = (bf16_t*)take((size_t)DDm * NK  * 2);
    bf16_t* wot  = (bf16_t*)take((size_t)NQ  * DDm * 2);
    bf16_t* sgt  = (bf16_t*)take((size_t)DDm * IIm * 2);
    bf16_t* sut  = (bf16_t*)take((size_t)DDm * IIm * 2);
    bf16_t* sdt  = (bf16_t*)take((size_t)IIm * DDm * 2);
    bf16_t* egt  = (bf16_t*)take((size_t)EEx * DDm * IIm * 2);
    bf16_t* eut  = (bf16_t*)take((size_t)EEx * DDm * IIm * 2);
    bf16_t* edt  = (bf16_t*)take((size_t)EEx * IIm * DDm * 2);
    bf16_t* hinb = (bf16_t*)take(T * DDm * 2);
    float*  qraw = (float*) take(T * NQ * 4);
    float*  kraw = (float*) take(T * NK * 4);
    float*  vraw = (float*) take(T * NK * 4);
    bf16_t* qb   = (bf16_t*)take(T * NQ * 2);
    bf16_t* ktb  = (bf16_t*)take((size_t)BB * KVH * HDD * SSQ * 2);
    bf16_t* vb   = (bf16_t*)take((size_t)BB * KVH * SSQ * HDD * 2);
    bf16_t* attb = (bf16_t*)take(T * NQ * 2);
    float*  h1   = (float*) take(T * DDm * 4);
    bf16_t* mb   = (bf16_t*)take(T * DDm * 2);
    float*  wd   = (float*) take(T * EEx * 4);
    bf16_t* gbuf = (bf16_t*)take(T * IIm * 2);
    bf16_t* ubuf = (bf16_t*)take(T * IIm * 2);
    float*  ybuf = (float*) take(T * DDm * 4);

    auto cdiv = [](size_t a, size_t b) -> unsigned { return (unsigned)((a + b - 1) / b); };

    // ---- weight conversions (transposed bf16) ----
    k_conv_t<<<cdiv((size_t)NQ  * DDm, 256), 256, 0, stream>>>(wq, wqt, NQ,  DDm);
    k_conv_t<<<cdiv((size_t)NK  * DDm, 256), 256, 0, stream>>>(wk, wkt, NK,  DDm);
    k_conv_t<<<cdiv((size_t)NK  * DDm, 256), 256, 0, stream>>>(wv, wvt, NK,  DDm);
    k_conv_t<<<cdiv((size_t)DDm * NQ,  256), 256, 0, stream>>>(wo, wot, DDm, NQ);
    k_conv_t<<<cdiv((size_t)IIm * DDm, 256), 256, 0, stream>>>(sgw, sgt, IIm, DDm);
    k_conv_t<<<cdiv((size_t)IIm * DDm, 256), 256, 0, stream>>>(suw, sut, IIm, DDm);
    k_conv_t<<<cdiv((size_t)DDm * IIm, 256), 256, 0, stream>>>(sdw, sdt, DDm, IIm);
    for (int e = 0; e < EEx; ++e) {
        k_conv_t<<<cdiv((size_t)IIm * DDm, 256), 256, 0, stream>>>(
            egw + (size_t)e * IIm * DDm, egt + (size_t)e * DDm * IIm, IIm, DDm);
        k_conv_t<<<cdiv((size_t)IIm * DDm, 256), 256, 0, stream>>>(
            euw + (size_t)e * IIm * DDm, eut + (size_t)e * DDm * IIm, IIm, DDm);
        k_conv_t<<<cdiv((size_t)DDm * IIm, 256), 256, 0, stream>>>(
            edw + (size_t)e * DDm * IIm, edt + (size_t)e * IIm * DDm, DDm, IIm);
    }

    auto gemm = [&](const bf16_t* A, const bf16_t* Bt, void* O, const float* Res,
                    const float* Scale, int ss, int M, int N, int K, int epi) {
        dim3 g((unsigned)(N / 64), (unsigned)(M / 128));
        k_gemm<<<g, 256, 0, stream>>>(A, Bt, O, Res, Scale, ss, M, N, K, epi);
    };

    // ---- attention branch ----
    k_rmsnorm<<<(unsigned)T, 32, 0, stream>>>(x, anw, hinb, DDm);

    gemm(hinb, wqt, qraw, nullptr, nullptr, 0, (int)T, NQ, DDm, EPI_F32);
    gemm(hinb, wkt, kraw, nullptr, nullptr, 0, (int)T, NK, DDm, EPI_F32);
    gemm(hinb, wvt, vraw, nullptr, nullptr, 0, (int)T, NK, DDm, EPI_F32);

    k_rope_q<<<(unsigned)(T * HHn), 32, 0, stream>>>(qraw, qnw, cosT, sinT, qb, 0.125f);
    k_rope_k<<<(unsigned)(T * KVH), 32, 0, stream>>>(kraw, knw, cosT, sinT, ktb);
    k_vconv<<<cdiv((size_t)BB * SSQ * KVH * HDD, 256), 256, 0, stream>>>(vraw, vb);

    k_attn<<<(unsigned)(BB * HHn * (SSQ / 16)), 32, 0, stream>>>(qb, ktb, vb, sinks, attb);

    gemm(attb, wot, h1, x, nullptr, 0, (int)T, DDm, NQ, EPI_ADDF32);   // h1 = x + attn@wo^T

    // ---- MoE branch ----
    k_rmsnorm<<<(unsigned)T, 32, 0, stream>>>(h1, pnw, mb, DDm);
    k_gate<<<(unsigned)T, 32, 0, stream>>>(mb, gatew, wd);
    k_zero<<<cdiv(T * DDm, 256), 256, 0, stream>>>(ybuf, T * DDm);

    for (int e = 0; e < EEx; ++e) {
        const bf16_t* eg = egt + (size_t)e * DDm * IIm;
        const bf16_t* eu = eut + (size_t)e * DDm * IIm;
        const bf16_t* ed = edt + (size_t)e * IIm * DDm;
        gemm(mb, eg, gbuf, nullptr, nullptr, 0, (int)T, IIm, DDm, EPI_GCLIP);
        gemm(mb, eu, ubuf, nullptr, nullptr, 0, (int)T, IIm, DDm, EPI_UCLIP);
        k_silu_mul<<<cdiv(T * IIm, 256), 256, 0, stream>>>(gbuf, ubuf, gbuf, T * IIm);
        gemm(gbuf, ed, ybuf, nullptr, wd + e, EEx, (int)T, DDm, IIm, EPI_SACC);
    }
    // shared FFN (weight 1.0)
    gemm(mb, sgt, gbuf, nullptr, nullptr, 0, (int)T, IIm, DDm, EPI_GCLIP);
    gemm(mb, sut, ubuf, nullptr, nullptr, 0, (int)T, IIm, DDm, EPI_UCLIP);
    k_silu_mul<<<cdiv(T * IIm, 256), 256, 0, stream>>>(gbuf, ubuf, gbuf, T * IIm);
    gemm(gbuf, sdt, ybuf, nullptr, nullptr, 0, (int)T, DDm, IIm, EPI_SACC);

    // ---- final residual ----
    k_addout<<<cdiv(T * DDm, 256), 256, 0, stream>>>(h1, ybuf, out, T * DDm);
}
